// aggregator_event_61856118997742
// MI455X (gfx1250) — matrix operand order repaired and checked by reading, not compiled, not run on hardware
//
#include <hip/hip_runtime.h>

// Problem constants (match reference)
#define NN      100000   // nodes   (divisible by 80)
#define NE      500000   // edges   (divisible by 80)
#define NRELS   500
#define HDIM    256
#define ODIM    128
#define MT      5        // 16-row M-tiles per block (80 rows) -> 5x B-fragment reuse
#define ROWS    (MT * 16)

typedef __attribute__((ext_vector_type(16))) __bf16 v16bf;
typedef __attribute__((ext_vector_type(8)))  float  v8f;

union Frag16 {
  v16bf v;
  unsigned short u[16];
  uint4 q[2];
};

__device__ __forceinline__ unsigned short f2bf(float f) {
  unsigned int x = __float_as_uint(f);
  unsigned int r = x + 0x7FFFu + ((x >> 16) & 1u);   // round-to-nearest-even
  return (unsigned short)(r >> 16);
}
__device__ __forceinline__ float bf2f(unsigned short h) {
  return __uint_as_float(((unsigned int)h) << 16);
}

// A fragment (16x32 bf16, M = lane&15): per ISA layout, half=0 lanes hold
// K = [0..7] and [16..23]; half=1 lanes hold K = [8..15] and [24..31].
__device__ __forceinline__ v16bf load_a_frag(const unsigned short* rowBase, int half) {
  Frag16 f;
  f.q[0] = *reinterpret_cast<const uint4*>(rowBase + 8 * half);
  f.q[1] = *reinterpret_cast<const uint4*>(rowBase + 16 + 8 * half);
  return f.v;
}
// B fragment (32x16 bf16): lane L holds column N = L&15; K across elements in the same order as A.
__device__ __forceinline__ v16bf load_b_frag(const unsigned short* W, int ldb,
                                             int kb, int col0, int lane) {
  const int hb = lane >> 4;
  const unsigned short* p = W + (size_t)(kb * 32) * ldb + col0 + (lane & 15);
  Frag16 f;
#pragma unroll
  for (int s = 0; s < 16; ++s) {
    int k = (s < 8) ? (8 * hb + s) : (16 + 8 * hb + (s - 8));
    f.u[s] = p[(size_t)k * ldb];
  }
  return f.v;
}

// ---------------- small prep kernels ----------------

__global__ void k_convert_bf16(const float* __restrict__ src,
                               unsigned short* __restrict__ dst, int n) {
  int i = blockIdx.x * blockDim.x + threadIdx.x;
  if (i < n) dst[i] = f2bf(src[i]);
}

__global__ void k_degrees(const int* __restrict__ esrc, const int* __restrict__ edst,
                          float* __restrict__ degO, float* __restrict__ degI, int E) {
  int i = blockIdx.x * blockDim.x + threadIdx.x;
  if (i >= E) return;
  atomicAdd(&degO[esrc[i]], 1.0f);
  atomicAdd(&degI[edst[i]], 1.0f);
}

__global__ void k_norms(const float* __restrict__ degI, const float* __restrict__ degO,
                        float* __restrict__ invdeg, float* __restrict__ nsrc,
                        float* __restrict__ ndst, int n) {
  int i = blockIdx.x * blockDim.x + threadIdx.x;
  if (i >= n) return;
  float di = fmaxf(degI[i], 1.0f);
  float dg = fmaxf(degO[i], 1.0f);
  invdeg[i] = 1.0f / di;
  nsrc[i]   = rsqrtf(dg);
  ndst[i]   = rsqrtf(di);
}

// rel1[r,c] = bf16( (rel_embeds @ W_rel1)[r,c] )   (tiny: 500x128, K=256)
__global__ void k_rel1(const float* __restrict__ relE, const float* __restrict__ Wrel1,
                       unsigned short* __restrict__ rel1) {
  int i = blockIdx.x * blockDim.x + threadIdx.x;
  if (i >= NRELS * ODIM) return;
  int r = i / ODIM, c = i % ODIM;
  float acc = 0.0f;
  for (int k = 0; k < HDIM; ++k)
    acc += relE[r * HDIM + k] * Wrel1[k * ODIM + c];
  rel1[i] = f2bf(acc);
}

// h_bf[n,:] = bf16(ent_embeds[node_ids[n],:]); one thread per 8 features
__global__ void k_gather_h(const int* __restrict__ ids, const float* __restrict__ ent,
                           unsigned short* __restrict__ hout, int n) {
  int idx = blockIdx.x * blockDim.x + threadIdx.x;
  if (idx >= n * 32) return;
  int node = idx >> 5, c = idx & 31;
  const float* p = ent + (size_t)ids[node] * HDIM + c * 8;
  Frag16 f;
#pragma unroll
  for (int i = 0; i < 8; ++i) f.u[i] = f2bf(p[i]);
  *reinterpret_cast<uint4*>(hout + (size_t)node * HDIM + c * 8) = f.q[0];
}

// ---------------- shared WMMA core ------------------------------------------------
// No column loop: each wave owns CTW = NCOLS/128 column tiles (cols (wave+8j)*16)
// and processes them simultaneously, so no load is loop-invariant (no LICM spill).
// Per kb: CTW B fragments + MT A fragments -> CTW*MT WMMAs.
template <int K, int NCOLS>
__device__ __forceinline__ void wmma_block(const unsigned short* As,
                                           const unsigned short* __restrict__ Wb,
                                           int wave, int lane, int M, int half,
                                           v8f c[NCOLS / 128][MT]) {
  constexpr int CTW = NCOLS / 128;
#pragma unroll
  for (int kb = 0; kb < K / 32; ++kb) {
    v16bf b[CTW];
#pragma unroll
    for (int j = 0; j < CTW; ++j)
      b[j] = load_b_frag(Wb, NCOLS, kb, (wave + 8 * j) * 16, lane);
#pragma unroll
    for (int m = 0; m < MT; ++m) {
      v16bf a = load_a_frag(&As[(m * 16 + M) * K + kb * 32], half);
#pragma unroll
      for (int j = 0; j < CTW; ++j)
        c[j][m] = __builtin_amdgcn_wmma_f32_16x16x32_bf16(false, a, false, b[j],
                                                          (short)0, c[j][m],
                                                          false, false);
    }
  }
}

// ---------------- edge message: WMMA + atomic scatter (segment_sum) ----------------
// agg[dst] += (h[src] * relTab[type]) @ W ; 80 edges per block (5 M-tiles).
template <int K, int NCOLS>
__global__ void k_edge_msg(const unsigned short* __restrict__ hin,    // [NN,K] bf16
                           const unsigned short* __restrict__ relTab, // [NRELS,K] bf16
                           const unsigned short* __restrict__ Wb,     // [K,NCOLS] bf16
                           const int* __restrict__ esrc, const int* __restrict__ etype,
                           const int* __restrict__ edst,
                           float* __restrict__ agg) {
  constexpr int CTW = NCOLS / 128;
  __shared__ __align__(16) unsigned short As[ROWS * K];
  const int tid = threadIdx.x;
  const int lane = tid & 31, wave = tid >> 5;
  const int eBase = blockIdx.x * ROWS;

  // compose A tile (80 edges x K) = h[src] * rel[type], in bf16 (E % 80 == 0)
  const int CH = K / 8;
  for (int idx = tid; idx < ROWS * CH; idx += 256) {
    int e = idx / CH, c = idx % CH;
    int ge = eBase + e;
    int s = esrc[ge], t = etype[ge];
    Frag16 hf, rf, of;
    hf.q[0] = *reinterpret_cast<const uint4*>(hin + (size_t)s * K + c * 8);
    rf.q[0] = *reinterpret_cast<const uint4*>(relTab + (size_t)t * K + c * 8);
#pragma unroll
    for (int i = 0; i < 8; ++i)
      of.u[i] = f2bf(bf2f(hf.u[i]) * bf2f(rf.u[i]));
    *reinterpret_cast<uint4*>(&As[e * K + c * 8]) = of.q[0];
  }
  __syncthreads();

  const int M = lane & 15, half = lane >> 4;
  v8f c[CTW][MT] = {};
  wmma_block<K, NCOLS>(As, Wb, wave, lane, M, half, c);

#pragma unroll
  for (int j = 0; j < CTW; ++j) {
    const int col = (wave + 8 * j) * 16 + (lane & 15);
#pragma unroll
    for (int m = 0; m < MT; ++m) {
#pragma unroll
      for (int r = 0; r < 8; ++r) {
        int ge = eBase + m * 16 + r + 8 * half;
        int d = edst[ge];
        atomicAdd(&agg[(size_t)d * NCOLS + col], c[j][m][r]);
      }
    }
  }
}

// ---------------- fused self-loop: hout = relu(agg*invdeg + hin @ W) ----------------
// 80 nodes per block; A rows staged global->LDS with coalesced 16B copies.
template <int K, int NCOLS>
__global__ void k_self_fused(const unsigned short* __restrict__ hin, // [NN,K] bf16
                             const unsigned short* __restrict__ Wb,  // [K,NCOLS] bf16
                             const float* __restrict__ agg,          // [NN,NCOLS]
                             const float* __restrict__ invdeg,
                             unsigned short* __restrict__ hout) {    // [NN,NCOLS] bf16
  constexpr int CTW = NCOLS / 128;
  __shared__ __align__(16) unsigned short As[ROWS * K];
  const int tid = threadIdx.x, lane = tid & 31, wave = tid >> 5;
  const int nBase = blockIdx.x * ROWS;
  const int M = lane & 15, half = lane >> 4;

  const int CH = K / 8;
  for (int idx = tid; idx < ROWS * CH; idx += 256) {
    int r = idx / CH, c = idx % CH;
    *reinterpret_cast<uint4*>(&As[r * K + c * 8]) =
        *reinterpret_cast<const uint4*>(hin + (size_t)(nBase + r) * K + c * 8);
  }
  __syncthreads();

  v8f c[CTW][MT] = {};
  wmma_block<K, NCOLS>(As, Wb, wave, lane, M, half, c);

#pragma unroll
  for (int j = 0; j < CTW; ++j) {
    const int col = (wave + 8 * j) * 16 + (lane & 15);
#pragma unroll
    for (int m = 0; m < MT; ++m) {
#pragma unroll
      for (int r = 0; r < 8; ++r) {
        int n = nBase + m * 16 + r + 8 * half;
        float v = agg[(size_t)n * NCOLS + col] * invdeg[n] + c[j][m][r];
        hout[(size_t)n * NCOLS + col] = f2bf(fmaxf(v, 0.0f));
      }
    }
  }
}

// ---------------- SCConv scatter: agg3[dst] += h2[src]*n_src[src] ----------------
__global__ void k_sc_scatter(const unsigned short* __restrict__ h2, // [NN,256] bf16
                             const float* __restrict__ nsrc,
                             const int* __restrict__ esrc, const int* __restrict__ edst,
                             float* __restrict__ agg, int E) {
  long long idx = (long long)blockIdx.x * blockDim.x + threadIdx.x;
  if (idx >= (long long)E * 32) return;
  int e = (int)(idx >> 5), c = (int)(idx & 31);
  int s = esrc[e], d = edst[e];
  float sc = nsrc[s];
  Frag16 f;
  f.q[0] = *reinterpret_cast<const uint4*>(h2 + (size_t)s * HDIM + c * 8);
  float* ap = agg + (size_t)d * HDIM + c * 8;
#pragma unroll
  for (int i = 0; i < 8; ++i) atomicAdd(&ap[i], bf2f(f.u[i]) * sc);
}

// ---------------- final: out = relu((agg3 * n_dst) @ W_sc) ----------------
// A tile = bf16(agg3 * n_dst) built cooperatively in LDS once per block.
__global__ void k_sc_final(const float* __restrict__ agg,           // [NN,256]
                           const float* __restrict__ ndst,
                           const unsigned short* __restrict__ Wsc,  // [256,256] bf16
                           float* __restrict__ out) {
  constexpr int CTW = 2;  // 256 cols / (8 waves * 16)
  __shared__ __align__(16) unsigned short As[ROWS * HDIM];
  const int tid = threadIdx.x, lane = tid & 31, wave = tid >> 5;
  const int nBase = blockIdx.x * ROWS;
  const int M = lane & 15, half = lane >> 4;

  for (int idx = tid; idx < ROWS * 32; idx += 256) {
    int r = idx / 32, c = idx % 32;
    int n = nBase + r;
    float sc = ndst[n];
    const float* p = agg + (size_t)n * HDIM + c * 8;
    Frag16 f;
#pragma unroll
    for (int i = 0; i < 8; ++i) f.u[i] = f2bf(p[i] * sc);
    *reinterpret_cast<uint4*>(&As[r * HDIM + c * 8]) = f.q[0];
  }
  __syncthreads();

  v8f c[CTW][MT] = {};
  wmma_block<HDIM, HDIM>(As, Wsc, wave, lane, M, half, c);

#pragma unroll
  for (int j = 0; j < CTW; ++j) {
    const int col = (wave + 8 * j) * 16 + (lane & 15);
#pragma unroll
    for (int m = 0; m < MT; ++m) {
#pragma unroll
      for (int r = 0; r < 8; ++r) {
        int n = nBase + m * 16 + r + 8 * half;
        out[(size_t)n * HDIM + col] = fmaxf(c[j][m][r], 0.0f);
      }
    }
  }
}

// ---------------- host launcher ----------------

extern "C" void kernel_launch(void* const* d_in, const int* in_sizes, int n_in,
                              void* d_out, int out_size, void* d_ws, size_t ws_size,
                              hipStream_t stream) {
  const int*   node_ids  = (const int*)  d_in[0];
  const int*   edge_type = (const int*)  d_in[1];
  const int*   edge_src  = (const int*)  d_in[2];
  const int*   edge_dst  = (const int*)  d_in[3];
  const float* ent       = (const float*)d_in[4];
  const float* relE_f32  = (const float*)d_in[5];
  const float* W_msg1    = (const float*)d_in[6];
  const float* W_self1   = (const float*)d_in[7];
  const float* W_rel1    = (const float*)d_in[8];
  const float* W_msg2    = (const float*)d_in[9];
  const float* W_self2   = (const float*)d_in[10];
  // d_in[11] = W_rel2 — dead in the reference (e2 never used)
  const float* W_sc      = (const float*)d_in[12];
  float* out = (float*)d_out;

  // workspace carve-out (256B aligned)
  char* base = (char*)d_ws;
  size_t off = 0;
  auto alloc = [&](size_t bytes) -> void* {
    void* p = base + off;
    off += (bytes + 255) & ~(size_t)255;
    return p;
  };
  float* deg_in  = (float*)alloc(sizeof(float) * NN);
  float* deg_out = (float*)alloc(sizeof(float) * NN);
  float* inv_deg = (float*)alloc(sizeof(float) * NN);
  float* n_src   = (float*)alloc(sizeof(float) * NN);
  float* n_dst   = (float*)alloc(sizeof(float) * NN);
  unsigned short* h_bf   = (unsigned short*)alloc(2ULL * NN * HDIM); // reused as h2_bf
  unsigned short* h1_bf  = (unsigned short*)alloc(2ULL * NN * ODIM);
  float* agg1            = (float*)alloc(4ULL * NN * ODIM);
  float* agg2            = (float*)alloc(4ULL * NN * HDIM);          // reused as agg3
  unsigned short* relE_bf   = (unsigned short*)alloc(2ULL * NRELS * HDIM);
  unsigned short* rel1_bf   = (unsigned short*)alloc(2ULL * NRELS * ODIM);
  unsigned short* Wmsg1_bf  = (unsigned short*)alloc(2ULL * HDIM * ODIM);
  unsigned short* Wself1_bf = (unsigned short*)alloc(2ULL * HDIM * ODIM);
  unsigned short* Wmsg2_bf  = (unsigned short*)alloc(2ULL * ODIM * HDIM);
  unsigned short* Wself2_bf = (unsigned short*)alloc(2ULL * ODIM * HDIM);
  unsigned short* Wsc_bf    = (unsigned short*)alloc(2ULL * HDIM * HDIM);

  const int T = 256;
  auto cdiv = [](long long a, long long b) { return (int)((a + b - 1) / b); };

  // degrees + normalizations
  hipMemsetAsync(deg_in, 0, sizeof(float) * NN, stream);
  hipMemsetAsync(deg_out, 0, sizeof(float) * NN, stream);
  k_degrees<<<cdiv(NE, T), T, 0, stream>>>(edge_src, edge_dst, deg_out, deg_in, NE);
  k_norms<<<cdiv(NN, T), T, 0, stream>>>(deg_in, deg_out, inv_deg, n_src, n_dst, NN);

  // bf16 weight/table conversions (tiny)
  k_convert_bf16<<<cdiv(HDIM * ODIM, T), T, 0, stream>>>(W_msg1, Wmsg1_bf, HDIM * ODIM);
  k_convert_bf16<<<cdiv(HDIM * ODIM, T), T, 0, stream>>>(W_self1, Wself1_bf, HDIM * ODIM);
  k_convert_bf16<<<cdiv(ODIM * HDIM, T), T, 0, stream>>>(W_msg2, Wmsg2_bf, ODIM * HDIM);
  k_convert_bf16<<<cdiv(ODIM * HDIM, T), T, 0, stream>>>(W_self2, Wself2_bf, ODIM * HDIM);
  k_convert_bf16<<<cdiv(HDIM * HDIM, T), T, 0, stream>>>(W_sc, Wsc_bf, HDIM * HDIM);
  k_convert_bf16<<<cdiv(NRELS * HDIM, T), T, 0, stream>>>(relE_f32, relE_bf, NRELS * HDIM);
  k_rel1<<<cdiv(NRELS * ODIM, T), T, 0, stream>>>(relE_f32, W_rel1, rel1_bf);

  // h = bf16(ent_embeds[node_ids])
  k_gather_h<<<cdiv((long long)NN * 32, T), T, 0, stream>>>(node_ids, ent, h_bf, NN);

  // ---- CompGCN layer 1 ----
  hipMemsetAsync(agg1, 0, 4ULL * NN * ODIM, stream);
  k_edge_msg<HDIM, ODIM><<<NE / ROWS, T, 0, stream>>>(h_bf, relE_bf, Wmsg1_bf,
                                                      edge_src, edge_type, edge_dst,
                                                      agg1);
  k_self_fused<HDIM, ODIM><<<NN / ROWS, T, 0, stream>>>(h_bf, Wself1_bf, agg1, inv_deg,
                                                        h1_bf);

  // ---- CompGCN layer 2 ----
  hipMemsetAsync(agg2, 0, 4ULL * NN * HDIM, stream);
  k_edge_msg<ODIM, HDIM><<<NE / ROWS, T, 0, stream>>>(h1_bf, rel1_bf, Wmsg2_bf,
                                                      edge_src, edge_type, edge_dst,
                                                      agg2);
  unsigned short* h2_bf = h_bf;  // reuse (h no longer needed)
  k_self_fused<ODIM, HDIM><<<NN / ROWS, T, 0, stream>>>(h1_bf, Wself2_bf, agg2, inv_deg,
                                                        h2_bf);

  // ---- SCConv ----
  float* agg3 = agg2;            // reuse (agg2 consumed by k_self_fused)
  hipMemsetAsync(agg3, 0, 4ULL * NN * HDIM, stream);
  k_sc_scatter<<<cdiv((long long)NE * 32, T), T, 0, stream>>>(h2_bf, n_src, edge_src,
                                                              edge_dst, agg3, NE);
  k_sc_final<<<NN / ROWS, T, 0, stream>>>(agg3, n_dst, Wsc_bf, out);
}
